// G2P_64433099374646
// MI455X (gfx1250) — compile-verified, matrix operand order checked by reading, not run on hardware
//
#include <hip/hip_runtime.h>
#include <hip/hip_bf16.h>
#include <math.h>

// ---------------------------------------------------------------------------
// Seq2seq LSTM (2-layer enc + 2-layer greedy decoder) for MI455X / gfx1250.
// Fused LSTM gate GEMM: gates = x*Wih^T + h*Whh^T + (bih+bhh) on the WMMA
// bf16 path (v_wmma_f32_16x16x32_bf16), f32 accumulation.
// W tiles are staged into LDS with the CDNA5 async path
// (global_load_async_to_lds_b128 + s_wait_asynccnt), double-buffered so the
// fill of chunk i+1 overlaps the WMMA work on chunk i. A fragments stream
// directly from global (no intra-block reuse to exploit). B fragments are
// preloaded 4-wide per K-step so one dscnt wait covers 8 WMMAs.
// ---------------------------------------------------------------------------

typedef __bf16 bf16_t;
typedef __bf16 v16bf __attribute__((ext_vector_type(16)));
typedef __bf16 v8bf  __attribute__((ext_vector_type(8)));
typedef float  v8f   __attribute__((ext_vector_type(8)));

#define BB   2048   // batch
#define TT   32     // seq len (src == tgt)
#define DD   512    // model dim
#define FD   2048   // 4*D (gate dim)
#define LL   2      // layers
#define DTGT 80     // target vocab

#define KC      64              // K-chunk staged per iteration
#define LDS_STR 72              // row stride in bf16 elements (144B, padded:
                                // conflict-free banks, 16B aligned)
#define LDS_BUF (64 * LDS_STR)  // one W buffer: 64 rows

__device__ __forceinline__ float sigm(float x) { return 1.0f / (1.0f + __expf(-x)); }

// Load one lane's 16-element bf16 fragment for the 16x16x32 WMMA:
// per CDNA5 ISA, a lane holds two contiguous 8-element K chunks
// ({kbase..kbase+7} and {kbase+16..kbase+23}) -> two 16-byte loads.
__device__ __forceinline__ v16bf frag_from_row(const bf16_t* row, int k) {
  union { v16bf v; v8bf h[2]; } u;
  u.h[0] = *(const v8bf*)(row + k);
  u.h[1] = *(const v8bf*)(row + k + 16);
  return u.v;
}

__device__ __forceinline__ v8f wmma_bf16(v16bf a, v16bf b, v8f c) {
  return __builtin_amdgcn_wmma_f32_16x16x32_bf16(false, a, false, b,
                                                 (short)0, c, false, false);
}

// CDNA5 async global->LDS copy, 16B per lane, tracked by ASYNCcnt.
__device__ __forceinline__ void async_load16(unsigned lds_addr, const void* gaddr) {
  asm volatile("global_load_async_to_lds_b128 %0, %1, off"
               :: "v"(lds_addr), "v"((unsigned long long)(uintptr_t)gaddr)
               : "memory");
}
__device__ __forceinline__ void wait_async0() {
  asm volatile("s_wait_asynccnt 0x0" ::: "memory");
}

// ---------------------------------------------------------------------------
// Fused gate GEMM:
//   Out[M,N] = A1[M,K]*W1[N,K]^T + A2[M,K]*W2[N,K]^T + bias[N]
// Block = 256 threads (8 waves), block tile 256x64; each wave computes a
// 32x64 strip (2 A-frags x 4 B-frags, 8 WMMAs per K-step). The 64xKC W chunk
// is loaded ONCE per block into LDS (async), shared by all 8 waves.
// Requires M%256==0, N%64==0, K%KC==0 (true for all call sites here).
// ---------------------------------------------------------------------------
__global__ __launch_bounds__(256)
void lstm_gates_wmma(const bf16_t* __restrict__ A1,   // x input   (M x K)
                     const bf16_t* __restrict__ W1,   // Wih       (N x K)
                     const bf16_t* __restrict__ A2,   // h input   (M x K)
                     const bf16_t* __restrict__ W2,   // Whh       (N x K)
                     const float*  __restrict__ bias, // N (or null)
                     float* __restrict__ Out,
                     int M, int N, int K) {
  __shared__ __align__(16) bf16_t smemW[2 * LDS_BUF];

  const int tid  = threadIdx.x;
  const int wave = tid >> 5;
  const int lane = tid & 31;
  const int m0 = blockIdx.x * 256 + wave * 32;
  const int n0 = blockIdx.y * 64;

  const int half  = lane >> 4;    // 0: lanes 0-15, 1: lanes 16-31
  const int sub   = lane & 15;
  const int kfrag = half * 8;     // lane's k-chunk base within each 32-wide step

  const int nchunk = 2 * (K / KC);          // two passes (W1 then W2)
  const unsigned lds_base = (unsigned)(uintptr_t)&smemW[0];

  // Issue the async fill of chunk i into LDS buffer (i&1).
  // 64 rows x 128B = 512 x 16B slots; 256 threads issue 2 each.
  auto issue_chunk = [&](int i) {
    const bf16_t* Wsrc = (i < (nchunk >> 1)) ? W1 : W2;
    const int kc = (i % (K / KC)) * KC;
    const unsigned lbase = lds_base + (unsigned)((i & 1) * LDS_BUF * 2);
#pragma unroll
    for (int j = 0; j < 2; ++j) {
      const int s = tid + j * 256;
      const int r = s >> 3;        // W row within strip (0..63)
      const int c = s & 7;         // 16B chunk within row (0..7)
      const unsigned laddr = lbase + (unsigned)(r * (LDS_STR * 2) + c * 16);
      const char* g = (const char*)Wsrc +
                      (((size_t)(n0 + r) * K + kc) * 2 + (size_t)c * 16);
      async_load16(laddr, g);
    }
  };

  v8f acc[2][4];
#pragma unroll
  for (int nt = 0; nt < 4; ++nt) {
    const float bv = bias ? bias[n0 + nt * 16 + sub] : 0.0f;
#pragma unroll
    for (int mt = 0; mt < 2; ++mt)
#pragma unroll
      for (int r = 0; r < 8; ++r) acc[mt][nt][r] = bv;
  }

  issue_chunk(0);
  wait_async0();
  __syncthreads();

  for (int i = 0; i < nchunk; ++i) {
    if (i + 1 < nchunk) issue_chunk(i + 1);   // overlap fill with compute

    const bf16_t* Ap = (i < (nchunk >> 1)) ? A1 : A2;
    const int kc = (i % (K / KC)) * KC;
    const bf16_t* Ar0 = Ap + (size_t)(m0 + sub) * K + kc;
    const bf16_t* Ar1 = Ar0 + (size_t)16 * K;
    const bf16_t* bbuf = &smemW[(i & 1) * LDS_BUF];

#pragma unroll
    for (int kl = 0; kl < KC; kl += 32) {
      const v16bf a0 = frag_from_row(Ar0, kl + kfrag);
      const v16bf a1 = frag_from_row(Ar1, kl + kfrag);
      // Preload all 4 B fragments into distinct registers: the 4 ds_load_b128
      // pairs issue as one clause, one dscnt wait covers the 8 WMMAs below.
      v16bf bfrag[4];
#pragma unroll
      for (int nt = 0; nt < 4; ++nt)
        bfrag[nt] = frag_from_row(bbuf + (nt * 16 + sub) * LDS_STR, kl + kfrag);
#pragma unroll
      for (int nt = 0; nt < 4; ++nt) {
        acc[0][nt] = wmma_bf16(a0, bfrag[nt], acc[0][nt]);
        acc[1][nt] = wmma_bf16(a1, bfrag[nt], acc[1][nt]);
      }
    }

    if (i + 1 < nchunk) wait_async0();  // next buffer landed
    __syncthreads();                    // all waves done with current buffer
  }

  // C/D layout: lane group selects rows {0..7} / {8..15} within each 16-row tile
  const int rb = m0 + half * 8;
#pragma unroll
  for (int mt = 0; mt < 2; ++mt)
#pragma unroll
    for (int nt = 0; nt < 4; ++nt)
#pragma unroll
      for (int r = 0; r < 8; ++r)
        Out[(size_t)(rb + mt * 16 + r) * N + (n0 + nt * 16 + sub)] =
            acc[mt][nt][r];
}

// ---------------------------------------------------------------------------
// LSTM pointwise: gates[B,4D] (i|f|g|o) -> update c, h; emit bf16 h (and
// optionally overwrite the time-major sequence buffer for the next layer).
// ---------------------------------------------------------------------------
__global__ __launch_bounds__(256)
void lstm_pointwise(const float* __restrict__ gates,
                    float* __restrict__ c,
                    float* __restrict__ h,
                    bf16_t* __restrict__ hb,
                    bf16_t* __restrict__ xseq) {
  const int idx = blockIdx.x * blockDim.x + threadIdx.x;   // over B*D
  if (idx >= BB * DD) return;
  const int b = idx / DD;
  const int d = idx - b * DD;
  const float* g = gates + (size_t)b * FD;
  const float ig = sigm(g[d]);
  const float fg = sigm(g[DD + d]);
  const float gg = tanhf(g[2 * DD + d]);
  const float og = sigm(g[3 * DD + d]);
  const float cn = fg * c[idx] + ig * gg;
  const float hn = og * tanhf(cn);
  c[idx] = cn;
  h[idx] = hn;
  hb[idx] = (bf16_t)hn;
  if (xseq) xseq[idx] = (bf16_t)hn;
}

// fp32 -> bf16 tensor convert
__global__ __launch_bounds__(256)
void cvt_bf16(const float* __restrict__ in, bf16_t* __restrict__ out, int n) {
  const int i = blockIdx.x * blockDim.x + threadIdx.x;
  if (i < n) out[i] = (bf16_t)in[i];
}

// out[i] = a[i] + b[i]  (combined bias bih+bhh)
__global__ __launch_bounds__(256)
void bias_sum(const float* __restrict__ a, const float* __restrict__ b,
              float* __restrict__ out, int n) {
  const int i = blockIdx.x * blockDim.x + threadIdx.x;
  if (i < n) out[i] = a[i] + b[i];
}

// X[(t*B + b)*D + d] = bf16(enc_emb[src[b*T + t]*D + d])   (time-major)
__global__ __launch_bounds__(256)
void embed_src(const int* __restrict__ src, const float* __restrict__ emb,
               bf16_t* __restrict__ X) {
  const int idx = blockIdx.x * blockDim.x + threadIdx.x;   // over T*B*D
  if (idx >= TT * BB * DD) return;
  const int d = idx % DD;
  const int r = idx / DD;
  const int b = r % BB;
  const int t = r / BB;
  const int tok = src[b * TT + t];
  X[idx] = (bf16_t)emb[(size_t)tok * DD + d];
}

// prev[b*D + d] = bf16(dec_emb[tgt[b*T + 0]*D + d])
__global__ __launch_bounds__(256)
void embed_prev0(const int* __restrict__ tgt, const float* __restrict__ emb,
                 bf16_t* __restrict__ prev) {
  const int idx = blockIdx.x * blockDim.x + threadIdx.x;   // over B*D
  if (idx >= BB * DD) return;
  const int b = idx / DD;
  const int d = idx - b * DD;
  const int tok = tgt[b * TT];
  prev[idx] = (bf16_t)emb[(size_t)tok * DD + d];
}

// ---------------------------------------------------------------------------
// Projection + argmax + embedding feedback. One block per batch row.
// logits[n] = h . postW[n] + post_b[n]; out[b,t,:]=logits; prev=dec_emb[argmax].
// ---------------------------------------------------------------------------
__global__ __launch_bounds__(128)
void proj_argmax_embed(const bf16_t* __restrict__ h,      // B x D
                       const bf16_t* __restrict__ Wp,     // DTGT x D
                       const float*  __restrict__ bp,     // DTGT
                       const bf16_t* __restrict__ demb,   // DTGT x D
                       float* __restrict__ out,           // B x T x DTGT
                       bf16_t* __restrict__ prev,         // B x D
                       int t) {
  const int b = blockIdx.x;
  __shared__ bf16_t hrow[DD];
  __shared__ float  logits[DTGT];
  __shared__ int    pred;

  for (int d = threadIdx.x; d < DD; d += blockDim.x)
    hrow[d] = h[(size_t)b * DD + d];
  __syncthreads();

  for (int n = threadIdx.x; n < DTGT; n += blockDim.x) {
    const bf16_t* w = Wp + (size_t)n * DD;
    float s = bp[n];
    for (int d = 0; d < DD; ++d) s += (float)hrow[d] * (float)w[d];
    logits[n] = s;
    out[((size_t)b * TT + t) * DTGT + n] = s;
  }
  __syncthreads();

  if (threadIdx.x == 0) {
    int best = 0;
    float bv = logits[0];
    for (int n = 1; n < DTGT; ++n)
      if (logits[n] > bv) { bv = logits[n]; best = n; }   // first-max tie-break
    pred = best;
  }
  __syncthreads();

  for (int d = threadIdx.x; d < DD; d += blockDim.x)
    prev[(size_t)b * DD + d] = demb[(size_t)pred * DD + d];
}

// ---------------------------------------------------------------------------
// Host orchestration
// ---------------------------------------------------------------------------
static inline void run_cell_gemm(const bf16_t* x, const bf16_t* Wih,
                                 const bf16_t* h, const bf16_t* Whh,
                                 const float* bias, float* gates,
                                 hipStream_t s) {
  dim3 grid(BB / 256, FD / 64);
  lstm_gates_wmma<<<grid, 256, 0, s>>>(x, Wih, h, Whh, bias, gates, BB, FD, DD);
}

extern "C" void kernel_launch(void* const* d_in, const int* in_sizes, int n_in,
                              void* d_out, int out_size, void* d_ws, size_t ws_size,
                              hipStream_t stream) {
  (void)in_sizes; (void)n_in; (void)out_size; (void)ws_size;

  const int*   src      = (const int*)d_in[0];
  const int*   tgt      = (const int*)d_in[1];
  const float* enc_emb  = (const float*)d_in[2];
  const float* dec_emb  = (const float*)d_in[3];
  const float* enc_Wih  = (const float*)d_in[4];
  const float* enc_Whh  = (const float*)d_in[5];
  const float* enc_bih  = (const float*)d_in[6];
  const float* enc_bhh  = (const float*)d_in[7];
  const float* dec_Wih  = (const float*)d_in[8];
  const float* dec_Whh  = (const float*)d_in[9];
  const float* dec_bih  = (const float*)d_in[10];
  const float* dec_bhh  = (const float*)d_in[11];
  const float* post_W   = (const float*)d_in[12];
  const float* post_b   = (const float*)d_in[13];
  float* out = (float*)d_out;

  // ---- workspace layout (~115 MB) ----
  char* p = (char*)d_ws;
  auto alloc = [&](size_t bytes) -> char* {
    char* r = p;
    p += (bytes + 255) & ~(size_t)255;
    return r;
  };
  const size_t WSZ = (size_t)LL * FD * DD;        // elements per (L,4D,D) tensor
  bf16_t* encWih_b = (bf16_t*)alloc(WSZ * 2);
  bf16_t* encWhh_b = (bf16_t*)alloc(WSZ * 2);
  bf16_t* decWih_b = (bf16_t*)alloc(WSZ * 2);
  bf16_t* decWhh_b = (bf16_t*)alloc(WSZ * 2);
  bf16_t* postW_b  = (bf16_t*)alloc((size_t)DTGT * DD * 2);
  bf16_t* demb_b   = (bf16_t*)alloc((size_t)DTGT * DD * 2);
  float*  enc_bsum = (float*)alloc((size_t)LL * FD * 4);
  float*  dec_bsum = (float*)alloc((size_t)LL * FD * 4);
  bf16_t* X        = (bf16_t*)alloc((size_t)TT * BB * DD * 2); // time-major seq
  float*  gates    = (float*)alloc((size_t)BB * FD * 4);
  float*  h_f32    = (float*)alloc((size_t)LL * BB * DD * 4);
  float*  c_f32    = (float*)alloc((size_t)LL * BB * DD * 4);
  bf16_t* h_b      = (bf16_t*)alloc((size_t)LL * BB * DD * 2);
  bf16_t* prev_b   = (bf16_t*)alloc((size_t)BB * DD * 2);

  const int CT = 256;
  auto nblk = [](size_t n, int t) { return (unsigned)((n + t - 1) / t); };

  // ---- weight / bias preprocessing (recomputed every call: deterministic) ----
  cvt_bf16<<<nblk(WSZ, CT), CT, 0, stream>>>(enc_Wih, encWih_b, (int)WSZ);
  cvt_bf16<<<nblk(WSZ, CT), CT, 0, stream>>>(enc_Whh, encWhh_b, (int)WSZ);
  cvt_bf16<<<nblk(WSZ, CT), CT, 0, stream>>>(dec_Wih, decWih_b, (int)WSZ);
  cvt_bf16<<<nblk(WSZ, CT), CT, 0, stream>>>(dec_Whh, decWhh_b, (int)WSZ);
  cvt_bf16<<<nblk((size_t)DTGT * DD, CT), CT, 0, stream>>>(post_W, postW_b, DTGT * DD);
  cvt_bf16<<<nblk((size_t)DTGT * DD, CT), CT, 0, stream>>>(dec_emb, demb_b, DTGT * DD);
  bias_sum<<<nblk((size_t)LL * FD, CT), CT, 0, stream>>>(enc_bih, enc_bhh, enc_bsum, LL * FD);
  bias_sum<<<nblk((size_t)LL * FD, CT), CT, 0, stream>>>(dec_bih, dec_bhh, dec_bsum, LL * FD);

  // initial hidden state = 0 (re-zeroed each call; graph-capture safe)
  hipMemsetAsync(h_f32, 0, (size_t)LL * BB * DD * 4, stream);
  hipMemsetAsync(c_f32, 0, (size_t)LL * BB * DD * 4, stream);
  hipMemsetAsync(h_b, 0, (size_t)LL * BB * DD * 2, stream);

  // source embeddings -> time-major X (overwritten layer by layer with h_t)
  embed_src<<<nblk((size_t)TT * BB * DD, CT), CT, 0, stream>>>(src, enc_emb, X);

  const unsigned pw_blocks = nblk((size_t)BB * DD, CT);

  // ---------------- encoder ----------------
  for (int l = 0; l < LL; ++l) {
    const bf16_t* Wih = encWih_b + (size_t)l * FD * DD;
    const bf16_t* Whh = encWhh_b + (size_t)l * FD * DD;
    const float*  bs  = enc_bsum + (size_t)l * FD;
    float*  hl = h_f32 + (size_t)l * BB * DD;
    float*  cl = c_f32 + (size_t)l * BB * DD;
    bf16_t* hb = h_b   + (size_t)l * BB * DD;
    for (int t = 0; t < TT; ++t) {
      bf16_t* xt = X + (size_t)t * BB * DD;
      // gates = x_t.Wih^T + h.Whh^T + (bih+bhh)  (fused, one kernel)
      run_cell_gemm(xt, Wih, hb, Whh, bs, gates, stream);
      // pointwise update; write h_t back into X[t] as next layer's input
      bf16_t* xnext = (l + 1 < LL) ? xt : nullptr;
      lstm_pointwise<<<pw_blocks, CT, 0, stream>>>(gates, cl, hl, hb, xnext);
    }
  }
  // encoder final (h, c) are already the decoder's initial state in h_*/c_*.

  // decoder initial input: dec_emb[tgt[:,0]]
  embed_prev0<<<pw_blocks, CT, 0, stream>>>(tgt, dec_emb, prev_b);

  // ---------------- decoder (greedy, sequential) ----------------
  for (int t = 0; t < TT; ++t) {
    for (int l = 0; l < LL; ++l) {
      const bf16_t* Wih = decWih_b + (size_t)l * FD * DD;
      const bf16_t* Whh = decWhh_b + (size_t)l * FD * DD;
      const float*  bs  = dec_bsum + (size_t)l * FD;
      const bf16_t* xin = (l == 0) ? prev_b : (h_b + (size_t)(l - 1) * BB * DD);
      float*  hl = h_f32 + (size_t)l * BB * DD;
      float*  cl = c_f32 + (size_t)l * BB * DD;
      bf16_t* hb = h_b   + (size_t)l * BB * DD;
      run_cell_gemm(xin, Wih, hb, Whh, bs, gates, stream);
      lstm_pointwise<<<pw_blocks, CT, 0, stream>>>(gates, cl, hl, hb, nullptr);
    }
    // logits, argmax, feedback embedding
    proj_argmax_embed<<<BB, 128, 0, stream>>>(
        h_b + (size_t)(LL - 1) * BB * DD, postW_b, post_b, demb_b, out, prev_b, t);
  }
}
